// MultiHeadAttention_52982716563683
// MI455X (gfx1250) — matrix operand order repaired and checked by reading, not compiled, not run on hardware
//
#include <hip/hip_runtime.h>
#include <hip/hip_bf16.h>

// ---------------------------------------------------------------------------
// MHA forward for gfx1250 (MI455X). Compute-bound (~32 GFLOP vs ~45 MB traffic
// at 23.3 TB/s), so all three matmul stages run on v_wmma_f32_16x16x32_f16
// (f16 in, f32 accumulate). V is produced pre-transposed so flash attention
// needs no LDS staging for K or V at all (both B-fragments are contiguous
// 32B/lane global loads); only P does a wave-private LDS C->A relayout,
// which needs no barriers (per-wave LDS ordering).
// ---------------------------------------------------------------------------

#define D_MODEL 768
#define N_HEADS 12
#define HEAD_DIM 64
#define BATCH 2
#define SEQ 2048
#define NTOK (BATCH * SEQ)          // 4096
#define QKV_N (3 * D_MODEL)         // 2304

typedef _Float16 v4h  __attribute__((ext_vector_type(4)));
typedef _Float16 v8h  __attribute__((ext_vector_type(8)));
typedef _Float16 v16h __attribute__((ext_vector_type(16)));
typedef float    v4f  __attribute__((ext_vector_type(4)));
typedef float    v8f  __attribute__((ext_vector_type(8)));

union U16 { v16h v; v8h h[2]; };

static __device__ inline v8f wmma_f16(const v16h& a, const v16h& b, const v8f& c) {
  return __builtin_amdgcn_wmma_f32_16x16x32_f16(false, a, false, b, (short)0, c,
                                                false, false);
}

static __device__ inline float halfgrp_max(float v) {
  v = fmaxf(v, __shfl_xor(v, 1, 32));
  v = fmaxf(v, __shfl_xor(v, 2, 32));
  v = fmaxf(v, __shfl_xor(v, 4, 32));
  v = fmaxf(v, __shfl_xor(v, 8, 32));
  return v;
}
static __device__ inline float halfgrp_sum(float v) {
  v += __shfl_xor(v, 1, 32);
  v += __shfl_xor(v, 2, 32);
  v += __shfl_xor(v, 4, 32);
  v += __shfl_xor(v, 8, 32);
  return v;
}

// ---------------------------------------------------------------------------
// Kernel 1: QKV projection. C[4096,2304] = X[4096,768] @ W[768,2304] + bias.
// 256 threads = 8 waves; tile 128x128; each wave computes 16x128 (8 accums).
// Epilogue scatters Q,K -> [B,H,T,64] and V -> transposed [B,H,64,T] (f16),
// with 1/sqrt(Dh) folded into Q. N-tile is 128-aligned so `which`/head-base
// are block-uniform scalars (no per-lane div/mod by 768).
// ---------------------------------------------------------------------------
__global__ __launch_bounds__(256)
void mha_qkv_gemm(const float* __restrict__ X, const float* __restrict__ W,
                  const float* __restrict__ bias,
                  _Float16* __restrict__ Qh, _Float16* __restrict__ Kh,
                  _Float16* __restrict__ VT) {
  __shared__ __align__(32) _Float16 As[128][40];   // [m][k]  (pad 32->40)
  __shared__ __align__(32) _Float16 Bs[128][48];   // [n][k]  (pad 32->48)

  const int tid  = threadIdx.x;
  const int wave = tid >> 5;
  const int lane = tid & 31;
  const int half = lane >> 4;
  const int ln   = lane & 15;
  const int m0   = blockIdx.y * 128;   // token base
  const int n0   = blockIdx.x * 128;   // output-column base

  v8f acc[8];
#pragma unroll
  for (int i = 0; i < 8; ++i) acc[i] = v8f{0.f};

  for (int k0 = 0; k0 < D_MODEL; k0 += 32) {
    __syncthreads();
    if (k0 + 32 < D_MODEL) {  // pull next tiles into near caches (WGP scope)
      __builtin_prefetch(&X[(long)(m0 + (tid >> 1)) * D_MODEL + k0 + 32], 0, 3);
      __builtin_prefetch(&W[(long)(k0 + 32 + (tid >> 4)) * QKV_N + n0], 0, 3);
    }
    // ---- A tile: 128x32 f32 -> f16.  4 x (b128 load -> cvt_pk -> b64 store).
    {
      v4f t[4];
#pragma unroll
      for (int i = 0; i < 4; ++i) {
        int vec = i * 256 + tid;            // 1024 float4 chunks
        int m = vec >> 3, kk = (vec & 7) * 4;
        t[i] = *(const v4f*)&X[(long)(m0 + m) * D_MODEL + k0 + kk];
      }
#pragma unroll
      for (int i = 0; i < 4; ++i) {
        int vec = i * 256 + tid;
        int m = vec >> 3, kk = (vec & 7) * 4;
        v4h h = {(_Float16)t[i].x, (_Float16)t[i].y,
                 (_Float16)t[i].z, (_Float16)t[i].w};
        *(v4h*)&As[m][kk] = h;
      }
    }
    // ---- B tile: Bs[n][kk] = W[k0+kk][n0+n]; gather 4 k per lane (each load
    //      coalesced over n), pack to one b64 LDS store.
    {
#pragma unroll
      for (int i = 0; i < 4; ++i) {
        int task = i * 256 + tid;           // 128 n * 8 k-groups
        int n = task & 127, g = task >> 7;
        const float* wp = &W[(long)(k0 + g * 4) * QKV_N + n0 + n];
        float w0 = wp[0];
        float w1 = wp[QKV_N];
        float w2 = wp[2 * QKV_N];
        float w3 = wp[3 * QKV_N];
        v4h h = {(_Float16)w0, (_Float16)w1, (_Float16)w2, (_Float16)w3};
        *(v4h*)&Bs[n][g * 4] = h;
      }
    }
    __syncthreads();

    // A fragment: row = wave*16+ln; k = half*16+0..15
    U16 a;
    a.h[0] = *(const v8h*)&As[wave * 16 + ln][half * 16];
    a.h[1] = *(const v8h*)&As[wave * 16 + ln][half * 16 + 8];
    // Grouped B-fragment loads: 8 ds_load_b128 clause -> 4 back-to-back WMMAs
#pragma unroll
    for (int grp = 0; grp < 2; ++grp) {
      U16 bf[4];
#pragma unroll
      for (int j = 0; j < 4; ++j)
        bf[j].v = *(const v16h*)&Bs[(grp * 4 + j) * 16 + ln][half * 16];
#pragma unroll
      for (int j = 0; j < 4; ++j)
        acc[grp * 4 + j] = wmma_f16(a.v, bf[j].v, acc[grp * 4 + j]);
    }
  }

  // ---- Epilogue. 768 % 128 == 0 -> `which` & head-base uniform per block.
  const int which = n0 / D_MODEL;                 // 0=Q 1=K 2=V
  const int hbase = (n0 - which * D_MODEL) >> 6;  // uniform head base
#pragma unroll
  for (int nt = 0; nt < 8; ++nt) {
    const int h = hbase + (nt >> 2);
    const int d = (nt & 3) * 16 + ln;
    const float bb = bias[n0 + nt * 16 + ln];
#pragma unroll
    for (int r = 0; r < 8; ++r) {
      int rowg = m0 + wave * 16 + r + half * 8;
      int b = rowg >> 11, t = rowg & (SEQ - 1);
      float v = acc[nt][r] + bb;
      long bhead = (long)(b * N_HEADS + h);
      if (which == 0)
        Qh[(bhead * SEQ + t) * HEAD_DIM + d] = (_Float16)(v * 0.125f);
      else if (which == 1)
        Kh[(bhead * SEQ + t) * HEAD_DIM + d] = (_Float16)v;
      else
        VT[(bhead * HEAD_DIM + d) * SEQ + t] = (_Float16)v;  // transposed
    }
  }
}

// ---------------------------------------------------------------------------
// Kernel 2: causal flash attention. grid = (T/64, B*H), 128 threads = 4 waves.
// Q fragments register-resident; K and V^T B-fragments are contiguous 32B
// global loads (no LDS staging, no barriers). P relayout C->A via
// wave-private LDS (per-wave DS ordering is architectural).
// ---------------------------------------------------------------------------
__global__ __launch_bounds__(128)
void mha_attention(const _Float16* __restrict__ Qh,
                   const _Float16* __restrict__ Kh,
                   const _Float16* __restrict__ VT,
                   _Float16* __restrict__ ctx) {
  __shared__ __align__(32) _Float16 Pbuf[4][16][72];   // wave-private P tiles

  const int qt   = blockIdx.x;
  const int bh   = blockIdx.y;
  const int tid  = threadIdx.x;
  const int wave = tid >> 5;
  const int lane = tid & 31;
  const int half = lane >> 4;
  const int ln   = lane & 15;

  const _Float16* Qp  = Qh + (long)bh * SEQ * HEAD_DIM;
  const _Float16* Kp  = Kh + (long)bh * SEQ * HEAD_DIM;
  const _Float16* Vtp = VT + (long)bh * HEAD_DIM * SEQ;

  const int qrow0 = qt * 64 + wave * 16;

  U16 qa[2];
  const _Float16* qrow = Qp + (long)(qrow0 + ln) * HEAD_DIM;
#pragma unroll
  for (int f = 0; f < 2; ++f) {
    qa[f].h[0] = *(const v8h*)(qrow + f * 32 + half * 16);
    qa[f].h[1] = *(const v8h*)(qrow + f * 32 + half * 16 + 8);
  }

  float m_i[8], l_i[8];
#pragma unroll
  for (int r = 0; r < 8; ++r) { m_i[r] = -1e30f; l_i[r] = 0.f; }
  v8f o[4];
#pragma unroll
  for (int i = 0; i < 4; ++i) o[i] = v8f{0.f};

  for (int kb = 0; kb <= qt; ++kb) {
    if (kb < qt) {  // prefetch next key/value block into near caches
      __builtin_prefetch(Kp + (long)((kb + 1) * 64 + (tid & 63)) * HEAD_DIM, 0, 3);
      __builtin_prefetch(Vtp + (long)(tid & 63) * SEQ + (kb + 1) * 64, 0, 3);
    }

    // ---- S = Q K^T (scale folded into Q); K B-frag contiguous from global.
    v8f s[4];
#pragma unroll
    for (int i = 0; i < 4; ++i) s[i] = v8f{0.f};
#pragma unroll
    for (int nt = 0; nt < 4; ++nt) {
      const _Float16* krow = Kp + (long)(kb * 64 + nt * 16 + ln) * HEAD_DIM;
      U16 b0, b1;
      b0.v = *(const v16h*)(krow + half * 16);
      b1.v = *(const v16h*)(krow + 32 + half * 16);
      s[nt] = wmma_f16(qa[0].v, b0.v, s[nt]);
      s[nt] = wmma_f16(qa[1].v, b1.v, s[nt]);
    }

    // ---- causal mask (diagonal block only)
    if (kb == qt) {
#pragma unroll
      for (int nt = 0; nt < 4; ++nt)
#pragma unroll
        for (int r = 0; r < 8; ++r)
          if (kb * 64 + nt * 16 + ln > qrow0 + r + half * 8)
            s[nt][r] = -1e30f;
    }

    // ---- online softmax; rows live in 16-lane half-groups
    float sc[8];
#pragma unroll
    for (int r = 0; r < 8; ++r) {
      float mx = fmaxf(fmaxf(s[0][r], s[1][r]), fmaxf(s[2][r], s[3][r]));
      mx = halfgrp_max(mx);
      float mnew = fmaxf(m_i[r], mx);
      sc[r] = __expf(m_i[r] - mnew);
      float rs = 0.f;
#pragma unroll
      for (int nt = 0; nt < 4; ++nt) {
        float p = __expf(s[nt][r] - mnew);
        s[nt][r] = p;
        rs += p;
      }
      rs = halfgrp_sum(rs);
      l_i[r] = l_i[r] * sc[r] + rs;
      m_i[r] = mnew;
    }
#pragma unroll
    for (int nt = 0; nt < 4; ++nt)
#pragma unroll
      for (int r = 0; r < 8; ++r) o[nt][r] *= sc[r];

    // ---- P: C-layout -> wave-private LDS -> A-layout fragments
#pragma unroll
    for (int nt = 0; nt < 4; ++nt)
#pragma unroll
      for (int r = 0; r < 8; ++r)
        Pbuf[wave][r + half * 8][nt * 16 + ln] = (_Float16)s[nt][r];

    U16 pa[2];
#pragma unroll
    for (int f = 0; f < 2; ++f) {
      pa[f].h[0] = *(const v8h*)&Pbuf[wave][ln][f * 32 + half * 16];
      pa[f].h[1] = *(const v8h*)&Pbuf[wave][ln][f * 32 + half * 16 + 8];
    }

    // ---- O += P @ V; V^T B-frag contiguous from global (keys along T)
#pragma unroll
    for (int nt = 0; nt < 4; ++nt) {
      const _Float16* vcol = Vtp + (long)(nt * 16 + ln) * SEQ + kb * 64;
      U16 vb0, vb1;
      vb0.v = *(const v16h*)(vcol + half * 16);
      vb1.v = *(const v16h*)(vcol + 32 + half * 16);
      o[nt] = wmma_f16(pa[0].v, vb0.v, o[nt]);
      o[nt] = wmma_f16(pa[1].v, vb1.v, o[nt]);
    }
  }

  // ---- normalize + write ctx [B,T,768] f16
  const int b = bh / N_HEADS, h = bh - b * N_HEADS;
#pragma unroll
  for (int nt = 0; nt < 4; ++nt)
#pragma unroll
    for (int r = 0; r < 8; ++r) {
      int q = qrow0 + r + half * 8;
      int d = nt * 16 + ln;
      ctx[((long)(b * SEQ + q)) * D_MODEL + h * HEAD_DIM + d] =
          (_Float16)(o[nt][r] / l_i[r]);
    }
}

// ---------------------------------------------------------------------------
// Kernel 3: output projection. out[4096,768] = ctx[4096,768] @ Wout + bout.
// A tile is already f16 -> staged with gfx1250 async global->LDS copies
// (ASYNCcnt-tracked, s_wait_asynccnt). Builtin signature per hipcc: param 0
// is int4 in AS(1) ("__device__"), param 1 the LDS-side pointer in AS(3).
// ---------------------------------------------------------------------------
#if defined(__has_builtin)
#if __has_builtin(__builtin_amdgcn_global_load_async_to_lds_b128)
#define USE_ASYNC_LDS 1
#endif
#endif

typedef int i4v __attribute__((vector_size(16)));
typedef __attribute__((address_space(1))) i4v* gbl_i4p;
typedef __attribute__((address_space(3))) i4v* lds_i4p;

__global__ __launch_bounds__(256)
void mha_out_gemm(const _Float16* __restrict__ A, const float* __restrict__ W,
                  const float* __restrict__ bias, float* __restrict__ out) {
  __shared__ __align__(32) _Float16 As[128][48];   // [m][k] pad 32->48 (16B rows)
  __shared__ __align__(32) _Float16 Bs[128][48];   // [n][k]

  const int tid  = threadIdx.x;
  const int wave = tid >> 5;
  const int lane = tid & 31;
  const int half = lane >> 4;
  const int ln   = lane & 15;
  const int m0   = blockIdx.y * 128;
  const int n0   = blockIdx.x * 128;

  v8f acc[8];
#pragma unroll
  for (int i = 0; i < 8; ++i) acc[i] = v8f{0.f};

  for (int k0 = 0; k0 < D_MODEL; k0 += 32) {
    __syncthreads();
    // ---- A tile: raw f16 copy, 512 x 16B chunks (async -> LDS on gfx1250)
#pragma unroll
    for (int i = 0; i < 2; ++i) {
      int chunk = i * 256 + tid;            // 4 chunks per 64B row-slice
      int m = chunk >> 2, c8 = (chunk & 3) * 8;
      const _Float16* g = A + (long)(m0 + m) * D_MODEL + k0 + c8;
#if USE_ASYNC_LDS
      __builtin_amdgcn_global_load_async_to_lds_b128(
          (gbl_i4p)(__attribute__((address_space(1))) void*)g,
          (lds_i4p)(__attribute__((address_space(3))) void*)&As[m][c8], 0, 0);
#else
      *(v8h*)&As[m][c8] = *(const v8h*)g;
#endif
    }
    // ---- B tile: gather 4 k per lane from Wout (coalesced over n), pack b64
#pragma unroll
    for (int i = 0; i < 4; ++i) {
      int task = i * 256 + tid;
      int n = task & 127, g = task >> 7;
      const float* wp = &W[(long)(k0 + g * 4) * D_MODEL + n0 + n];
      float w0 = wp[0];
      float w1 = wp[D_MODEL];
      float w2 = wp[2 * D_MODEL];
      float w3 = wp[3 * D_MODEL];
      v4h h = {(_Float16)w0, (_Float16)w1, (_Float16)w2, (_Float16)w3};
      *(v4h*)&Bs[n][g * 4] = h;
    }
#if USE_ASYNC_LDS
    asm volatile("s_wait_asynccnt 0x0" ::: "memory");
#endif
    __syncthreads();

    U16 a;
    a.h[0] = *(const v8h*)&As[wave * 16 + ln][half * 16];
    a.h[1] = *(const v8h*)&As[wave * 16 + ln][half * 16 + 8];
#pragma unroll
    for (int grp = 0; grp < 2; ++grp) {
      U16 bf[4];
#pragma unroll
      for (int j = 0; j < 4; ++j)
        bf[j].v = *(const v16h*)&Bs[(grp * 4 + j) * 16 + ln][half * 16];
#pragma unroll
      for (int j = 0; j < 4; ++j)
        acc[grp * 4 + j] = wmma_f16(a.v, bf[j].v, acc[grp * 4 + j]);
    }
  }

#pragma unroll
  for (int nt = 0; nt < 8; ++nt) {
    const float bb = bias[n0 + nt * 16 + ln];
#pragma unroll
    for (int r = 0; r < 8; ++r) {
      int rowg = m0 + wave * 16 + r + half * 8;
      out[(long)rowg * D_MODEL + n0 + nt * 16 + ln] = acc[nt][r] + bb;
    }
  }
}

// ---------------------------------------------------------------------------
extern "C" void kernel_launch(void* const* d_in, const int* in_sizes, int n_in,
                              void* d_out, int out_size, void* d_ws,
                              size_t ws_size, hipStream_t stream) {
  const float* x    = (const float*)d_in[0];
  // d_in[1] = causal mask (ignored: causality computed analytically)
  const float* Wqkv = (const float*)d_in[2];
  const float* bqkv = (const float*)d_in[3];
  const float* Wout = (const float*)d_in[4];
  const float* bout = (const float*)d_in[5];
  float* out        = (float*)d_out;

  // workspace (f16): Q [B,H,T,64], K [B,H,T,64], V^T [B,H,64,T], ctx [B,T,768]
  const size_t seg = (size_t)BATCH * N_HEADS * SEQ * HEAD_DIM * sizeof(_Float16);
  char* ws = (char*)d_ws;
  _Float16* Qh  = (_Float16*)(ws + 0 * seg);
  _Float16* Kh  = (_Float16*)(ws + 1 * seg);
  _Float16* VT  = (_Float16*)(ws + 2 * seg);
  _Float16* ctx = (_Float16*)(ws + 3 * seg);

  mha_qkv_gemm<<<dim3(QKV_N / 128, NTOK / 128), 256, 0, stream>>>(
      x, Wqkv, bqkv, Qh, Kh, VT);
  mha_attention<<<dim3(SEQ / 64, BATCH * N_HEADS), 128, 0, stream>>>(
      Qh, Kh, VT, ctx);
  mha_out_gemm<<<dim3(D_MODEL / 128, NTOK / 128), 256, 0, stream>>>(
      ctx, Wout, bout, out);
}
